// ContrastiveLoss_68496138437096
// MI455X (gfx1250) — matrix operand order repaired and checked by reading, not compiled, not run on hardware
//
#include <hip/hip_runtime.h>
#include <hip/hip_bf16.h>
#include <hip/hip_fp16.h>

typedef __attribute__((ext_vector_type(16))) _Float16 v16h;
typedef __attribute__((ext_vector_type(8)))  _Float16 v8h;
typedef __attribute__((ext_vector_type(8)))  float    v8f;

#define B_N   1024
#define C_N   256
#define HW_N  576
#define INV_T (1.0f / 0.07f)

// ---------------------------------------------------------------------------
// Kernel 1: global mean pool over H,W (24x24=576) + L2 normalize -> f16 row.
// One block per image; one wave per channel (float4 coalesced loads).
// 604 MB read total -> HBM-bound stage (~26 us at 23.3 TB/s).
// ---------------------------------------------------------------------------
__global__ __launch_bounds__(256) void pool_norm_kernel(
    const float* __restrict__ img, _Float16* __restrict__ outh) {
  __shared__ float mean[C_N];
  __shared__ float wsq[8];
  __shared__ float inv_norm;

  const int b    = blockIdx.x;
  const int lane = threadIdx.x & 31;
  const int wave = threadIdx.x >> 5;
  const float* base = img + (size_t)b * C_N * HW_N;

  for (int c = wave; c < C_N; c += 8) {
    const float4* p = (const float4*)(base + c * HW_N);
    float s = 0.f;
#pragma unroll
    for (int i = 0; i < 4; ++i) {           // 512 floats, coalesced float4
      float4 v = p[lane + 32 * i];
      s += v.x + v.y + v.z + v.w;
    }
    if (lane < 16) {                        // remaining 64 floats
      float4 v = p[128 + lane];
      s += v.x + v.y + v.z + v.w;
    }
#pragma unroll
    for (int off = 16; off > 0; off >>= 1) s += __shfl_xor(s, off, 32);
    if (lane == 0) mean[c] = s * (1.0f / (float)HW_N);
  }
  __syncthreads();

  const float v  = mean[threadIdx.x];
  float sq = v * v;
#pragma unroll
  for (int off = 16; off > 0; off >>= 1) sq += __shfl_xor(sq, off, 32);
  if (lane == 0) wsq[wave] = sq;
  __syncthreads();
  if (threadIdx.x == 0) {
    float t = 0.f;
#pragma unroll
    for (int i = 0; i < 8; ++i) t += wsq[i];
    inv_norm = 1.0f / fmaxf(sqrtf(t), 1e-12f);
  }
  __syncthreads();
  outh[(size_t)b * C_N + threadIdx.x] = (_Float16)(v * inv_norm);
}

// ---------------------------------------------------------------------------
// Kernel 2: L2 normalize text embeds [1024,256] -> f16 rows.
// ---------------------------------------------------------------------------
__global__ __launch_bounds__(256) void text_norm_kernel(
    const float* __restrict__ txt, _Float16* __restrict__ outh) {
  __shared__ float wsq[8];
  __shared__ float inv_norm;
  const int b    = blockIdx.x;
  const int lane = threadIdx.x & 31;
  const int wave = threadIdx.x >> 5;

  const float v = txt[(size_t)b * C_N + threadIdx.x];
  float sq = v * v;
#pragma unroll
  for (int off = 16; off > 0; off >>= 1) sq += __shfl_xor(sq, off, 32);
  if (lane == 0) wsq[wave] = sq;
  __syncthreads();
  if (threadIdx.x == 0) {
    float t = 0.f;
#pragma unroll
    for (int i = 0; i < 8; ++i) t += wsq[i];
    inv_norm = 1.0f / fmaxf(sqrtf(t), 1e-12f);
  }
  __syncthreads();
  outh[(size_t)b * C_N + threadIdx.x] = (_Float16)(v * inv_norm);
}

// ---------------------------------------------------------------------------
// Kernel 3: 16x16 tile of logits via v_wmma_f32_16x16x32_f16 (8 WMMAs, K=256),
// fused softplus-BCE, wave-reduced partial per tile (deterministic, no atomics).
//
// Fragment layouts used here (wave32; contiguous k, same order in A and B):
//  A (16x32 f16): lane L -> row (L&15); elems j = K (L>>4)*16+j
//  B (32x16 f16) = Tn^T: lane L -> col (L&15); elems j = K (L>>4)*16+j
//                 -> 16 contiguous f16 from a row of Tn (no transpose op needed)
//  C/D (16x16 f32): VGPR r -> M=(L>>4)*8+r, N=(L&15)
// ---------------------------------------------------------------------------
__global__ __launch_bounds__(32) void gemm_loss_kernel(
    const _Float16* __restrict__ An, const _Float16* __restrict__ Tn,
    const int* __restrict__ labels, float* __restrict__ partial) {
  const int tn = blockIdx.x;       // N tile (text)
  const int tm = blockIdx.y;       // M tile (image)
  const int lane = threadIdx.x;
  const int l15  = lane & 15;
  const int kh   = lane >> 4;

  const _Float16* arow = An + (size_t)(tm * 16 + l15) * C_N;
  const _Float16* brow = Tn + (size_t)(tn * 16 + l15) * C_N;

  v8f acc = {};
#pragma unroll
  for (int kk = 0; kk < C_N; kk += 32) {
    v16h a = *(const v16h*)(arow + kk + kh * 16);       // 32B aligned
    v16h bfrag = *(const v16h*)(brow + kk + kh * 16);   // 32B aligned
    acc = __builtin_amdgcn_wmma_f32_16x16x32_f16(
        /*neg_a=*/false, a, /*neg_b=*/false, bfrag,
        /*c_mod=*/(short)0, acc, /*reuse_a=*/false, /*reuse_b=*/false);
  }

  const int lab_col = labels[tn * 16 + l15];
  float s = 0.f;
#pragma unroll
  for (int r = 0; r < 8; ++r) {
    const int row = tm * 16 + kh * 8 + r;
    const float x = acc[r] * INV_T;
    const float z = (labels[row] == lab_col) ? 1.f : 0.f;
    // logaddexp(0,x) - x*z  ==  max(x,0) + log1p(exp(-|x|)) - x*z
    const float sp = fmaxf(x, 0.f) + log1pf(__expf(-fabsf(x)));
    s += sp - x * z;
  }
#pragma unroll
  for (int off = 16; off > 0; off >>= 1) s += __shfl_xor(s, off, 32);
  if (lane == 0) partial[tm * 64 + tn] = s;
}

// ---------------------------------------------------------------------------
// Kernel 4: fixed-order reduction of 4096 tile partials -> scalar mean loss.
// ---------------------------------------------------------------------------
__global__ __launch_bounds__(256) void reduce_kernel(
    const float* __restrict__ partial, float* __restrict__ out) {
  __shared__ float sm[256];
  float s = 0.f;
  for (int i = threadIdx.x; i < 4096; i += 256) s += partial[i];
  sm[threadIdx.x] = s;
  __syncthreads();
  for (int off = 128; off > 0; off >>= 1) {
    if (threadIdx.x < off) sm[threadIdx.x] += sm[threadIdx.x + off];
    __syncthreads();
  }
  if (threadIdx.x == 0) out[0] = sm[0] * (1.0f / (1024.0f * 1024.0f));
}

extern "C" void kernel_launch(void* const* d_in, const int* in_sizes, int n_in,
                              void* d_out, int out_size, void* d_ws, size_t ws_size,
                              hipStream_t stream) {
  const float* img    = (const float*)d_in[0];   // [1024,256,24,24] f32
  const float* txt    = (const float*)d_in[1];   // [1024,256] f32
  const int*   labels = (const int*)d_in[2];     // [1024] int

  _Float16* imgn    = (_Float16*)d_ws;                    // 512 KB
  _Float16* txtn    = imgn + (size_t)B_N * C_N;           // 512 KB
  float*    partial = (float*)(txtn + (size_t)B_N * C_N); // 16 KB

  pool_norm_kernel<<<B_N, 256, 0, stream>>>(img, imgn);
  text_norm_kernel<<<B_N, 256, 0, stream>>>(txt, txtn);
  gemm_loss_kernel<<<dim3(64, 64), 32, 0, stream>>>(imgn, txtn, labels, partial);
  reduce_kernel<<<1, 256, 0, stream>>>(partial, (float*)d_out);
}
